// Head_21990232556059
// MI455X (gfx1250) — matrix operand order repaired and checked by reading, not compiled, not run on hardware
//
#include <hip/hip_runtime.h>

// ---------------------------------------------------------------------------
// Causal single-head attention for MI455X (gfx1250, wave32, WMMA).
//   B=8, T=2048, C=1024, H=64.  fp32 in/out, f16 WMMA internally (f32 accum).
// Pass 0: one-time W f32->f16 transpose into WT[3*64 n][1024 k] (384 KB).
// Pass 1: q/k/v projections via v_wmma_f32_16x16x32_f16. A-frags converted
//         from global f32 x in registers; B-frags are direct b128 loads from
//         WT. No LDS, no barriers -> fully pipelineable K loop.
// Pass 2: flash attention, one wave per 16-row query tile, causal-limited
//         key loop, online softmax, WMMA for S=QK^T and O=PV, prefetch of
//         next key block (global_prefetch_b8).
// ---------------------------------------------------------------------------

typedef __attribute__((ext_vector_type(8)))  _Float16 v8h;
typedef __attribute__((ext_vector_type(16))) _Float16 v16h;
typedef __attribute__((ext_vector_type(8)))  float    v8f;

#define WMMA_F16(a, b, c) \
  __builtin_amdgcn_wmma_f32_16x16x32_f16(false, (a), false, (b), (short)0, (c), false, false)

static __device__ __forceinline__ v16h cat16(v8h lo, v8h hi) {
  return __builtin_shufflevector(lo, hi, 0, 1, 2, 3, 4, 5, 6, 7, 8, 9, 10, 11, 12, 13, 14, 15);
}
static __device__ __forceinline__ v8f zero8() {
  v8f z = {0.f, 0.f, 0.f, 0.f, 0.f, 0.f, 0.f, 0.f};
  return z;
}

#define ATT_B 8
#define ATT_T 2048
#define ATT_C 1024
#define ATT_H 64
#define LOG2E 1.44269504088896340736f

// ---------------------------------------------------------------------------
// Pass 0: WT[(mat*64 + n)*1024 + k] = (f16) W_mat[k*64 + n]
// 192 blocks x 256 threads, coalesced reads; one-time 768 KB -> negligible.
// ---------------------------------------------------------------------------
__global__ __launch_bounds__(256) void wconv_kernel(
    const float* __restrict__ Wq, const float* __restrict__ Wk,
    const float* __restrict__ Wv, _Float16* __restrict__ WT) {
  const int bx    = blockIdx.x;   // 0..191
  const int mat   = bx >> 6;
  const int kslab = bx & 63;      // 16 k-rows per block
  const float* W = (mat == 0) ? Wq : ((mat == 1) ? Wk : Wv);
#pragma unroll
  for (int it = 0; it < 4; ++it) {
    const int idx = threadIdx.x + it * 256;     // 0..1023
    const int k = kslab * 16 + (idx >> 6);
    const int n = idx & 63;
    WT[((mat << 6) + n) * ATT_C + k] = (_Float16)W[k * ATT_H + n];
  }
}

// ---------------------------------------------------------------------------
// Pass 1: projections. Grid: B*T/16 blocks x 128 threads (4 waves).
// Wave w owns output n-tile w for all three matrices. No LDS.
// ---------------------------------------------------------------------------
__global__ __launch_bounds__(128) void proj_qkv_kernel(
    const float* __restrict__ x, const _Float16* __restrict__ WT,
    _Float16* __restrict__ qh, _Float16* __restrict__ kh,
    _Float16* __restrict__ vT) {
  constexpr int T = ATT_T, C = ATT_C, H = ATT_H;

  const int tid  = threadIdx.x;
  const int lane = tid & 31;
  const int wave = tid >> 5;      // 0..3 -> output n-tile
  const int g    = lane >> 4;     // lane half-group
  const int ln   = lane & 15;

  const long g0 = (long)blockIdx.x * 16;       // global row in [B*T)
  const int  b  = (int)(g0 / T);
  const int  t0 = (int)(g0 % T);

  // A-frag row for this lane (A layout: M = ln for both half-groups)
  const float* xrow = &x[(g0 + ln) * C];
  // B-frag rows (B layout: N = ln + 16*g? no: N = ln, K = g*8 + e (e < 8), 16 + g*8 + e-8 (e >= 8))
  const _Float16* w0 = &WT[(0 * 64 + wave * 16 + ln) * C];
  const _Float16* w1 = &WT[(1 * 64 + wave * 16 + ln) * C];
  const _Float16* w2 = &WT[(2 * 64 + wave * 16 + ln) * C];

  v8f acc0 = zero8(), acc1 = zero8(), acc2 = zero8();

  for (int k0 = 0; k0 < C; k0 += 32) {
    // ---- A fragment: two runs of 8 consecutive-k f32 values per lane, cvt to f16.
    //      elements 0..7 -> k = k0 + g*8 + e ; 8..15 -> k = k0 + 16 + g*8 + e
    const float4 f0 = *(const float4*)&xrow[k0 + g * 8 + 0];
    const float4 f1 = *(const float4*)&xrow[k0 + g * 8 + 4];
    const float4 f2 = *(const float4*)&xrow[k0 + 16 + g * 8 + 0];
    const float4 f3 = *(const float4*)&xrow[k0 + 16 + g * 8 + 4];
    v16h a;
    a[0]  = (_Float16)f0.x; a[1]  = (_Float16)f0.y;
    a[2]  = (_Float16)f0.z; a[3]  = (_Float16)f0.w;
    a[4]  = (_Float16)f1.x; a[5]  = (_Float16)f1.y;
    a[6]  = (_Float16)f1.z; a[7]  = (_Float16)f1.w;
    a[8]  = (_Float16)f2.x; a[9]  = (_Float16)f2.y;
    a[10] = (_Float16)f2.z; a[11] = (_Float16)f2.w;
    a[12] = (_Float16)f3.x; a[13] = (_Float16)f3.y;
    a[14] = (_Float16)f3.z; a[15] = (_Float16)f3.w;

    // ---- B fragments: elements 0..7 -> K = k0 + g*8 + e ; 8..15 -> K = k0 + 16 + g*8 + e
    const int kk = k0 + g * 8;
    const v16h b0 = cat16(*(const v8h*)&w0[kk], *(const v8h*)&w0[kk + 16]);
    const v16h b1 = cat16(*(const v8h*)&w1[kk], *(const v8h*)&w1[kk + 16]);
    const v16h b2 = cat16(*(const v8h*)&w2[kk], *(const v8h*)&w2[kk + 16]);

    acc0 = WMMA_F16(a, b0, acc0);
    acc1 = WMMA_F16(a, b1, acc1);
    acc2 = WMMA_F16(a, b2, acc2);
  }

  // C/D layout: VGPR r -> M = r + 8*g, N = ln
#pragma unroll
  for (int r = 0; r < 8; ++r) {
    const int  m   = r + 8 * g;
    const int  h   = wave * 16 + ln;
    const long row = g0 + m;
    qh[row * H + h] = (_Float16)acc0[r];
    kh[row * H + h] = (_Float16)acc1[r];
    vT[((long)b * H + h) * (long)T + t0 + m] = (_Float16)acc2[r];  // transposed
  }
}

// ---------------------------------------------------------------------------
// Pass 2: flash attention. One wave (32 thr) per 16-row query tile.
// ---------------------------------------------------------------------------
__global__ __launch_bounds__(32) void flash_attn_kernel(
    const _Float16* __restrict__ qh, const _Float16* __restrict__ kh,
    const _Float16* __restrict__ vT, float* __restrict__ out) {
  constexpr int T = ATT_T, H = ATT_H;
  constexpr int LD = 40;
  const float scale = 0.03125f;  // C^-0.5 = 1024^-0.5 = 1/32

  __shared__ _Float16 Pbuf[16 * LD];  // C/D -> A-frag relayout buffer

  const int lane = threadIdx.x & 31;
  const int g = lane >> 4, ln = lane & 15;
  const int blk = blockIdx.x;          // 0 .. B*T/16-1
  const int b   = blk / (T / 16);
  const int tq  = blk % (T / 16);
  const int q0  = tq * 16;             // first query row (in-batch)
  const long gr = (long)b * T + q0;    // global query row base

  // Q as two resident A-fragments (h chunks 0..31, 32..63)
  v16h aq[2];
#pragma unroll
  for (int c = 0; c < 2; ++c) {
    const _Float16* p = &qh[(gr + ln) * H + c * 32 + g * 8];
    aq[c] = cat16(*(const v8h*)p, *(const v8h*)(p + 16));
  }

  float mstat[8], lstat[8];
  v8f o[4];
#pragma unroll
  for (int r = 0; r < 8; ++r) { mstat[r] = -1e30f; lstat[r] = 0.f; }
#pragma unroll
  for (int h = 0; h < 4; ++h) o[h] = zero8();

  const int nblocks = (q0 + 16 + 31) >> 5;  // causal limit, 32-key blocks
  for (int kb = 0; kb < nblocks; ++kb) {
    const int j = kb * 32;

    // ---- prefetch next key block's K and V lines (global_prefetch_b8)
    if (kb + 1 < nblocks) {
      const int jn = j + 32;
      __builtin_prefetch(&kh[((long)b * T + jn + lane) * H], 0, 3);
      __builtin_prefetch(&vT[((long)b * H + lane) * (long)T + jn], 0, 3);
      __builtin_prefetch(&vT[((long)b * H + 32 + lane) * (long)T + jn], 0, 3);
    }

    // ---- S = Q @ K^T  (two 16-col n-tiles, K-dim = H = 2 chunks)
    v8f s[2];
#pragma unroll
    for (int nn = 0; nn < 2; ++nn) {
      const long key = (long)b * T + j + nn * 16 + ln;  // B-frag col = key
      v8f acc = zero8();
#pragma unroll
      for (int c = 0; c < 2; ++c) {
        const _Float16* p = &kh[key * H + c * 32 + g * 8];
        const v16h bk = cat16(*(const v8h*)p, *(const v8h*)(p + 16));
        acc = WMMA_F16(aq[c], bk, acc);
      }
      s[nn] = acc;
    }

    // ---- causal mask + online softmax (row = 16 lanes of a half-group)
    float p0[8], p1[8];
#pragma unroll
    for (int r = 0; r < 8; ++r) {
      const int qrow = q0 + r + 8 * g;
      const float s0 = ((j + ln)      > qrow) ? -1e30f : s[0][r] * scale;
      const float s1 = ((j + 16 + ln) > qrow) ? -1e30f : s[1][r] * scale;
      float rv = fmaxf(s0, s1);
      rv = fmaxf(rv, __shfl_xor(rv, 1, 16));
      rv = fmaxf(rv, __shfl_xor(rv, 2, 16));
      rv = fmaxf(rv, __shfl_xor(rv, 4, 16));
      rv = fmaxf(rv, __shfl_xor(rv, 8, 16));
      const float mnew = fmaxf(mstat[r], rv);
      const float f  = exp2f((mstat[r] - mnew) * LOG2E);
      const float e0 = exp2f((s0 - mnew) * LOG2E);  // masked -> underflow to 0
      const float e1 = exp2f((s1 - mnew) * LOG2E);
      float rs = e0 + e1;
      rs += __shfl_xor(rs, 1, 16);
      rs += __shfl_xor(rs, 2, 16);
      rs += __shfl_xor(rs, 4, 16);
      rs += __shfl_xor(rs, 8, 16);
      lstat[r] = lstat[r] * f + rs;
      mstat[r] = mnew;
#pragma unroll
      for (int h = 0; h < 4; ++h) o[h][r] *= f;
      p0[r] = e0;
      p1[r] = e1;
    }

    // ---- relayout P (C/D layout) -> A-fragment via LDS
    __syncthreads();
#pragma unroll
    for (int r = 0; r < 8; ++r) {
      const int m = r + 8 * g;
      Pbuf[m * LD + ln]      = (_Float16)p0[r];
      Pbuf[m * LD + 16 + ln] = (_Float16)p1[r];
    }
    __syncthreads();
    const v8h plo = *(const v8h*)&Pbuf[ln * LD + g * 8];
    const v8h phi = *(const v8h*)&Pbuf[ln * LD + 16 + g * 8];
    const v16h pa = cat16(plo, phi);

    // ---- O += P @ V  (vT[B,H,T]: B-frag col = h, K contiguous in t)
#pragma unroll
    for (int ht = 0; ht < 4; ++ht) {
      const int h = ht * 16 + ln;
      const _Float16* p = &vT[((long)b * H + h) * (long)T + j + g * 8];
      const v16h bv = cat16(*(const v8h*)p, *(const v8h*)(p + 16));
      o[ht] = WMMA_F16(pa, bv, o[ht]);
    }
    __syncthreads();  // Pbuf reused next iteration
  }

  // ---- epilogue: O / l, fp32 out [B,T,H]
#pragma unroll
  for (int r = 0; r < 8; ++r) {
    const int m = r + 8 * g;
    const float inv = 1.0f / lstat[r];
#pragma unroll
    for (int ht = 0; ht < 4; ++ht) {
      out[(gr + m) * H + ht * 16 + ln] = o[ht][r] * inv;
    }
  }
}

// ---------------------------------------------------------------------------
extern "C" void kernel_launch(void* const* d_in, const int* in_sizes, int n_in,
                              void* d_out, int out_size, void* d_ws,
                              size_t ws_size, hipStream_t stream) {
  (void)in_sizes; (void)n_in; (void)out_size; (void)ws_size;
  const float* x  = (const float*)d_in[0];
  const float* Wq = (const float*)d_in[1];
  const float* Wk = (const float*)d_in[2];
  const float* Wv = (const float*)d_in[3];
  float* out = (float*)d_out;

  const size_t elems = (size_t)ATT_B * ATT_T * ATT_H;  // 1 Mi halves each
  _Float16* qh = (_Float16*)d_ws;
  _Float16* kh = qh + elems;
  _Float16* vT = kh + elems;
  _Float16* WT = vT + elems;  // 3*64*1024 halves; total ws ~6.4 MB

  wconv_kernel<<<192, 256, 0, stream>>>(Wq, Wk, Wv, WT);
  const int ntiles = ATT_B * ATT_T / 16;  // 1024
  proj_qkv_kernel<<<ntiles, 128, 0, stream>>>(x, WT, qh, kh, vT);
  flash_attn_kernel<<<ntiles, 32, 0, stream>>>(qh, kh, vT, out);
}
